// ARRNN_35648228557273
// MI455X (gfx1250) — compile-verified, matrix operand order checked by reading, not run on hardware
//
#include <hip/hip_runtime.h>
#include <stddef.h>

#define HD   256
#define BATCH 4096
#define TOBS 64
#define TUNOBS 32
#define IND  4

typedef __attribute__((ext_vector_type(16))) __bf16 v16bf;
typedef __attribute__((ext_vector_type(8)))  __bf16 v8bf;
typedef __attribute__((ext_vector_type(8)))  float  v8f;

static __device__ __forceinline__ v16bf ld_tile16(const __bf16* p) {
    // A/B operand packing per CDNA5 ISA 7.12.2 (16-bit, 16x32 tile):
    // elements 0..7  = K[k0 .. k0+7]   (VGPR0..3)
    // elements 8..15 = K[k0+16..k0+23] (VGPR4..7)
    union { v16bf v; v8bf h[2]; } u;
    u.h[0] = *(const v8bf*)(p);
    u.h[1] = *(const v8bf*)(p + 16);
    return u.v;
}

static __device__ __forceinline__ v8f wmma_bf16(v16bf a, v16bf b, v8f c) {
    return __builtin_amdgcn_wmma_f32_16x16x32_bf16(
        /*neg_a=*/false, a, /*neg_b=*/false, b,
        /*c_mod=*/(short)0, c, /*reuse_a=*/false, /*reuse_b=*/false);
}

static __device__ __forceinline__ float sigmoid_f(float x) {
    return 1.0f / (1.0f + __expf(-x));
}
static __device__ __forceinline__ float tanh_f(float x) {
    float e = __expf(2.0f * x);
    return 1.0f - 2.0f / (e + 1.0f);
}

// ---------------------------------------------------------------- utilities
__global__ void k_cvt_bf16(const float* __restrict__ in, __bf16* __restrict__ out, int n) {
    int i = blockIdx.x * 256 + threadIdx.x;
    if (i < n) out[i] = (__bf16)in[i];
}

// ---------------------------------------------------------------- stem MLP
// rows = TOBS*BATCH.  hidden1 = leaky_relu(W1 @ [x, dt] + b1)  (K=5, VALU)
// xstem = hidden1 @ W2^T + b2                                   (K=256, WMMA)
__global__ __launch_bounds__(128) void k_stem(
    const float* __restrict__ x_obs, const float* __restrict__ t_obs,
    const float* __restrict__ w1, const float* __restrict__ b1,
    const __bf16* __restrict__ w2bf, const float* __restrict__ b2,
    __bf16* __restrict__ xstem)
{
    __shared__ float  feat[64][8];
    __shared__ float  w1s[HD * 5];
    __shared__ float  b1s[HD];
    __shared__ __align__(16) __bf16 a_lds[64][264];   // 256 + 8 pad

    const int tid = threadIdx.x;
    const int rowBlk = blockIdx.x;   // 64 rows each, 4096 blocks
    const int colBlk = blockIdx.y;   // 64 cols each, 4 blocks

    for (int i = tid; i < HD * 5; i += 128) w1s[i] = w1[i];
    for (int i = tid; i < HD;     i += 128) b1s[i] = b1[i];

    if (tid < 64) {
        int r = rowBlk * 64 + tid;
        int t = r / BATCH, b = r % BATCH;
        const float* xr = x_obs + ((size_t)t * BATCH + b) * IND;
        feat[tid][0] = xr[0]; feat[tid][1] = xr[1];
        feat[tid][2] = xr[2]; feat[tid][3] = xr[3];
        float dt = 0.0f;
        if (t > 0) dt = t_obs[(size_t)t * BATCH + b] - t_obs[(size_t)(t - 1) * BATCH + b];
        feat[tid][4] = dt;
    }
    __syncthreads();

    for (int e = tid; e < 64 * HD; e += 128) {
        int r = e >> 8, c = e & 255;
        const float* wr = &w1s[c * 5];
        float v = b1s[c] + wr[0] * feat[r][0] + wr[1] * feat[r][1] + wr[2] * feat[r][2]
                         + wr[3] * feat[r][3] + wr[4] * feat[r][4];
        v = (v > 0.0f) ? v : 0.01f * v;           // leaky_relu, slope 0.01
        a_lds[r][c] = (__bf16)v;
    }
    __syncthreads();

    const int wave = tid >> 5, lane = tid & 31, lm = lane & 15;
    const int sel = (lane < 16) ? 0 : 8;
    const int arow = wave * 16 + lm;

    v8f acc[4] = {};
    for (int k0 = 0; k0 < HD; k0 += 32) {
        v16bf a = ld_tile16(&a_lds[arow][k0 + sel]);
        #pragma unroll
        for (int nt = 0; nt < 4; ++nt) {
            int j = colBlk * 64 + nt * 16 + lm;
            v16bf bm = ld_tile16(w2bf + (size_t)j * HD + k0 + sel);
            acc[nt] = wmma_bf16(a, bm, acc[nt]);
        }
    }

    const size_t rowBase = (size_t)rowBlk * 64 + wave * 16;
    const int moff = (lane < 16) ? 0 : 8;
    #pragma unroll
    for (int nt = 0; nt < 4; ++nt) {
        int j = colBlk * 64 + nt * 16 + lm;
        float bj = b2[j];
        #pragma unroll
        for (int v = 0; v < 8; ++v) {
            size_t row = rowBase + v + moff;
            xstem[row * HD + j] = (__bf16)(acc[nt][v] + bj);
        }
    }
}

// ---------------------------------------------------------------- GRU step
// h_new = (1-u)*n + u*h, u/r = sigmoid(gx+gh), n = tanh(nx + r*nh)
// r,z gates: gx and gh chained into one WMMA accumulator; n kept split.
// Wave tile: 32 rows x 32 cols (2 M-tiles x 2 N-tiles), so each B (weight)
// tile fetched once feeds two WMMAs -> 24 WMMAs per 32 b128 loads per K-step.
__global__ __launch_bounds__(128) void k_gru_step(
    const __bf16* __restrict__ x, const float* __restrict__ hin,
    const __bf16* __restrict__ hinbf, float* __restrict__ hout,
    __bf16* __restrict__ houtbf,
    const __bf16* __restrict__ wi, const __bf16* __restrict__ wh,
    const float* __restrict__ bi, const float* __restrict__ bh,
    float* __restrict__ zs)
{
    const int tid = threadIdx.x, wave = tid >> 5, lane = tid & 31, lm = lane & 15;
    const int sel = (lane < 16) ? 0 : 8;
    const int mBlk = blockIdx.x;   // 128 batch rows per WG (32 per wave)
    const int nBlk = blockIdx.y;   // 32 hidden cols

    const size_t rbase = (size_t)mBlk * 128 + wave * 32 + lm;
    const __bf16* xr[2] = { x     + rbase * HD, x     + (rbase + 16) * HD };
    const __bf16* hr[2] = { hinbf + rbase * HD, hinbf + (rbase + 16) * HD };

    v8f aR[2][2] = {}, aZ[2][2] = {}, aNX[2][2] = {}, aNH[2][2] = {};

    for (int k0 = 0; k0 < HD; k0 += 32) {
        v16bf ax[2], ah[2];
        #pragma unroll
        for (int mt = 0; mt < 2; ++mt) {
            ax[mt] = ld_tile16(xr[mt] + k0 + sel);
            ah[mt] = ld_tile16(hr[mt] + k0 + sel);
        }
        #pragma unroll
        for (int nt = 0; nt < 2; ++nt) {
            int j = nBlk * 32 + nt * 16 + lm;
            v16bf br_i = ld_tile16(wi + (size_t)(         j) * HD + k0 + sel);
            v16bf br_h = ld_tile16(wh + (size_t)(         j) * HD + k0 + sel);
            v16bf bz_i = ld_tile16(wi + (size_t)(HD     + j) * HD + k0 + sel);
            v16bf bz_h = ld_tile16(wh + (size_t)(HD     + j) * HD + k0 + sel);
            v16bf bn_i = ld_tile16(wi + (size_t)(2 * HD + j) * HD + k0 + sel);
            v16bf bn_h = ld_tile16(wh + (size_t)(2 * HD + j) * HD + k0 + sel);
            #pragma unroll
            for (int mt = 0; mt < 2; ++mt) {
                aR[mt][nt]  = wmma_bf16(ax[mt], br_i, aR[mt][nt]);
                aR[mt][nt]  = wmma_bf16(ah[mt], br_h, aR[mt][nt]);
                aZ[mt][nt]  = wmma_bf16(ax[mt], bz_i, aZ[mt][nt]);
                aZ[mt][nt]  = wmma_bf16(ah[mt], bz_h, aZ[mt][nt]);
                aNX[mt][nt] = wmma_bf16(ax[mt], bn_i, aNX[mt][nt]);
                aNH[mt][nt] = wmma_bf16(ah[mt], bn_h, aNH[mt][nt]);
            }
        }
    }

    const int moff = (lane < 16) ? 0 : 8;
    #pragma unroll
    for (int nt = 0; nt < 2; ++nt) {
        int j = nBlk * 32 + nt * 16 + lm;
        float bR  = bi[j] + bh[j];
        float bZ  = bi[HD + j] + bh[HD + j];
        float bNX = bi[2 * HD + j];
        float bNH = bh[2 * HD + j];
        #pragma unroll
        for (int mt = 0; mt < 2; ++mt) {
            #pragma unroll
            for (int v = 0; v < 8; ++v) {
                size_t row = (size_t)mBlk * 128 + wave * 32 + mt * 16 + v + moff;
                float r  = sigmoid_f(aR[mt][nt][v] + bR);
                float u  = sigmoid_f(aZ[mt][nt][v] + bZ);
                float n  = tanh_f(aNX[mt][nt][v] + bNX + r * (aNH[mt][nt][v] + bNH));
                float ho = hin[row * HD + j];
                float hn = (1.0f - u) * n + u * ho;
                hout[row * HD + j]   = hn;
                houtbf[row * HD + j] = (__bf16)hn;
                if (zs) zs[row * HD + j] = hn;
            }
        }
    }
}

// ---------------------------------------------------------------- head
__global__ __launch_bounds__(256) void k_head(
    const float* __restrict__ zs, const float* __restrict__ head_w,
    const float* __restrict__ head_b, float* __restrict__ xh)
{
    __shared__ float w[IND * HD];
    __shared__ float bb[IND];
    const int tid = threadIdx.x;
    for (int i = tid; i < IND * HD; i += 256) w[i] = head_w[i];
    if (tid < IND) bb[tid] = head_b[tid];
    __syncthreads();

    size_t r = (size_t)blockIdx.x * 256 + tid;     // TUNOBS*BATCH rows
    const float* zr = zs + r * HD;
    float a0 = bb[0], a1 = bb[1], a2 = bb[2], a3 = bb[3];
    for (int j = 0; j < HD; ++j) {
        float v = zr[j];
        a0 += v * w[j];
        a1 += v * w[HD + j];
        a2 += v * w[2 * HD + j];
        a3 += v * w[3 * HD + j];
    }
    float* o = xh + r * IND;
    o[0] = a0; o[1] = a1; o[2] = a2; o[3] = a3;
}

// ---------------------------------------------------------------- host
extern "C" void kernel_launch(void* const* d_in, const int* in_sizes, int n_in,
                              void* d_out, int out_size, void* d_ws, size_t ws_size,
                              hipStream_t stream) {
    (void)in_sizes; (void)n_in; (void)out_size; (void)ws_size;

    const float* x_obs   = (const float*)d_in[0];
    const float* t_obs   = (const float*)d_in[1];
    // d_in[2] = t_unobs (length only, unused)
    const float* stem_w1 = (const float*)d_in[3];
    const float* stem_b1 = (const float*)d_in[4];
    const float* stem_w2 = (const float*)d_in[5];
    const float* stem_b2 = (const float*)d_in[6];
    const float* gru_wi  = (const float*)d_in[7];
    const float* gru_wh  = (const float*)d_in[8];
    const float* gru_bi  = (const float*)d_in[9];
    const float* gru_bh  = (const float*)d_in[10];
    const float* head_w  = (const float*)d_in[11];
    const float* head_b  = (const float*)d_in[12];

    // workspace carve-up
    char* ws = (char*)d_ws;
    size_t o = 0;
    __bf16* wi_bf = (__bf16*)(ws + o); o += (size_t)3 * HD * HD * 2;         // 384 KB
    __bf16* wh_bf = (__bf16*)(ws + o); o += (size_t)3 * HD * HD * 2;         // 384 KB
    __bf16* w2_bf = (__bf16*)(ws + o); o += (size_t)HD * HD * 2;             // 128 KB
    float*  hf[2];
    hf[0] = (float*)(ws + o); o += (size_t)BATCH * HD * 4;                   // 4 MB
    hf[1] = (float*)(ws + o); o += (size_t)BATCH * HD * 4;                   // 4 MB
    __bf16* hb[2];
    hb[0] = (__bf16*)(ws + o); o += (size_t)BATCH * HD * 2;                  // 2 MB
    hb[1] = (__bf16*)(ws + o); o += (size_t)BATCH * HD * 2;                  // 2 MB
    __bf16* xstem = (__bf16*)(ws + o); o += (size_t)TOBS * BATCH * HD * 2;   // 128 MB

    // weight conversion f32 -> bf16
    {
        int n1 = 3 * HD * HD, n2 = HD * HD;
        k_cvt_bf16<<<(n1 + 255) / 256, 256, 0, stream>>>(gru_wi,  wi_bf, n1);
        k_cvt_bf16<<<(n1 + 255) / 256, 256, 0, stream>>>(gru_wh,  wh_bf, n1);
        k_cvt_bf16<<<(n2 + 255) / 256, 256, 0, stream>>>(stem_w2, w2_bf, n2);
    }

    // h0 = 0
    hipMemsetAsync(hf[0], 0, (size_t)BATCH * HD * 4, stream);
    hipMemsetAsync(hb[0], 0, (size_t)BATCH * HD * 2, stream);

    // stem for all observed timesteps
    k_stem<<<dim3(TOBS * BATCH / 64, HD / 64), 128, 0, stream>>>(
        x_obs, t_obs, stem_w1, stem_b1, w2_bf, stem_b2, xstem);

    float* x_hats  = (float*)d_out;                                   // [32,4096,4]
    float* zs_base = (float*)d_out + (size_t)TUNOBS * BATCH * IND;    // [32,4096,256]

    const dim3 ggrid(BATCH / 128, HD / 32);
    int cur = 0;

    // observed scan (last step also writes zs[0] = z_last)
    for (int t = 0; t < TOBS; ++t) {
        float* zsp = (t == TOBS - 1) ? zs_base : nullptr;
        k_gru_step<<<ggrid, 128, 0, stream>>>(
            xstem + (size_t)t * BATCH * HD,
            hf[cur], hb[cur], hf[cur ^ 1], hb[cur ^ 1],
            wi_bf, wh_bf, gru_bi, gru_bh, zsp);
        cur ^= 1;
    }

    // autoregressive scan: x_t = z = h (bf16 mirror of current h)
    for (int t = 1; t < TUNOBS; ++t) {
        k_gru_step<<<ggrid, 128, 0, stream>>>(
            hb[cur],
            hf[cur], hb[cur], hf[cur ^ 1], hb[cur ^ 1],
            wi_bf, wh_bf, gru_bi, gru_bh,
            zs_base + (size_t)t * BATCH * HD);
        cur ^= 1;
    }

    // head: x_hats = zs @ head_w^T + head_b
    k_head<<<(TUNOBS * BATCH) / 256, 256, 0, stream>>>(zs_base, head_w, head_b, x_hats);
}
